// LSTM_CRF_Model_37400575213954
// MI455X (gfx1250) — compile-verified, matrix operand order checked by reading, not run on hardware
//
#include <hip/hip_runtime.h>

typedef __attribute__((ext_vector_type(16))) _Float16 v16h;
typedef __attribute__((ext_vector_type(8)))  float    v8f;

#define DEV __device__ __forceinline__

constexpr int Ecst = 128, Hcst = 256, Tcst = 35, Bcst = 64, Lcst = 1024;
constexpr int G4H  = 4 * Hcst;   // 1024 (gates, PyTorch order i,f,g,o)
constexpr int KX   = Ecst / 32;  // 4 K-tiles from x
constexpr int KH   = Hcst / 32;  // 8 K-tiles from h
constexpr int KT   = KX + KH;    // 12
constexpr int NTL  = G4H / 16;   // 64 N-tiles
constexpr int START_TAG = 33;

DEV float sigmoidf(float x) { return 1.0f / (1.0f + __expf(-x)); }

// ISA 7.12.2 16-bit A/B fragment: lane = r + 16*hi, slot s in 0..15 holds
// K(s,hi) = 16*(s>=8) + 8*hi + (s&7).
DEV int frag_k(int s, int hi) { return ((s & 8) << 1) + 8 * hi + (s & 7); }

DEV v8f wmma16(v16h a, v16h b, v8f c) {
  return __builtin_amdgcn_wmma_f32_16x16x32_f16(false, a, false, b, (short)0, c,
                                                false, false);
}

// ---------------- x: embedding gather straight into A-fragment layout -------
// xpack[bt(4)][t(1024)][kt(4)][lane(32)][s(16)] f16
__global__ __launch_bounds__(256) void k_embed_pack(
    const int* __restrict__ tok, const float* __restrict__ emb,
    _Float16* __restrict__ xpack) {
  const int i    = blockIdx.x * 256 + threadIdx.x;  // < 4*1024*4*512 = 8.4M
  const int s    = i & 15;
  const int lane = (i >> 4) & 31;
  const int kt   = (i >> 9) & 3;
  const int t    = (i >> 11) & 1023;
  const int bt   = i >> 21;
  const int m    = lane & 15;
  const int hi   = lane >> 4;
  const int b    = bt * 16 + m;
  const int k    = kt * 32 + frag_k(s, hi);
  xpack[i] = (_Float16)emb[(size_t)tok[b * Lcst + t] * Ecst + k];
}

// ---------------- weights: pack into B-fragment layout ----------------------
// wpack[nt(64)][kt(12)][lane(32)][s(16)] f16, K fused = [E | H]
// clspack[nt(3)][kt(16)][lane(32)][s(16)] f16 (rows >= 35 zero)
__global__ __launch_bounds__(256) void k_pack_w(
    const float* __restrict__ wihf, const float* __restrict__ whhf,
    const float* __restrict__ bihf, const float* __restrict__ bhhf,
    const float* __restrict__ wihb, const float* __restrict__ whhb,
    const float* __restrict__ bihb, const float* __restrict__ bhhb,
    const float* __restrict__ clsw,
    _Float16* __restrict__ wpackf, _Float16* __restrict__ wpackb,
    _Float16* __restrict__ clspack, float* __restrict__ biasf,
    float* __restrict__ biasb) {
  const int i = blockIdx.x * 256 + threadIdx.x;  // grid covers 64*12*512
  {
    const int s    = i & 15;
    const int lane = (i >> 4) & 31;
    const int j    = i >> 9;        // nt*12 + kt
    const int kt   = j % KT;
    const int nt   = j / KT;
    const int n    = lane & 15;
    const int hi   = lane >> 4;
    const int fk   = frag_k(s, hi);
    const int row  = nt * 16 + n;
    if (kt < KX) {
      const int k = kt * 32 + fk;
      wpackf[i] = (_Float16)wihf[row * Ecst + k];
      wpackb[i] = (_Float16)wihb[row * Ecst + k];
    } else {
      const int k = (kt - KX) * 32 + fk;
      wpackf[i] = (_Float16)whhf[row * Hcst + k];
      wpackb[i] = (_Float16)whhb[row * Hcst + k];
    }
  }
  if (i < 3 * 16 * 512) {  // classifier pack
    const int s    = i & 15;
    const int lane = (i >> 4) & 31;
    const int j    = i >> 9;        // nt*16 + kt
    const int kt   = j & 15;
    const int nt   = j >> 4;
    const int n    = lane & 15;
    const int hi   = lane >> 4;
    const int row  = nt * 16 + n;
    const int k    = kt * 32 + frag_k(s, hi);
    clspack[i] = (_Float16)((row < Tcst) ? clsw[row * (2 * Hcst) + k] : 0.0f);
  }
  if (i < G4H) {
    biasf[i] = bihf[i] + bhhf[i];
    biasb[i] = bihb[i] + bhhb[i];
  }
}

// -------------------------------------------------------------- BiLSTM scan
// 8 blocks: blockIdx = dir*4 + batch_tile; 32 waves, wave w owns N-tiles 2w,2w+1.
// hpack output is in the classifier's A-fragment layout:
//   hpack[mt = bt*1024 + t][kslot(16) = dir*8 + h/32][lane][s]
__global__ __launch_bounds__(1024) void k_lstm(
    const _Float16* __restrict__ xpack,
    const _Float16* __restrict__ wpackf, const float* __restrict__ biasf,
    const _Float16* __restrict__ wpackb, const float* __restrict__ biasb,
    _Float16* __restrict__ hpack) {
  const int dir = blockIdx.x >> 2;
  const int bt  = blockIdx.x & 3;
  const _Float16* wpack = dir ? wpackb : wpackf;
  const float*    bias  = dir ? biasb : biasf;

  __shared__ _Float16 sh_hf[KH * 32 * 16];  //  8 KB, h in A-fragment layout
  __shared__ float    sh_c[16 * Hcst];      // 16 KB
  __shared__ _Float16 sh_g[16 * G4H];       // 32 KB, pre-activation gates

  const int tid  = threadIdx.x;
  const int wave = tid >> 5;
  const int lane = tid & 31;

  for (int i = tid; i < KH * 512; i += 1024) sh_hf[i] = (_Float16)0.0f;
  for (int i = tid; i < 16 * Hcst; i += 1024) sh_c[i] = 0.0f;
  for (int i = tid * 32; i < NTL * KT * 512; i += 1024 * 32)
    __builtin_prefetch(&wpack[i], 0, 3);
  __syncthreads();

  const int  nt0 = wave * 2, nt1 = wave * 2 + 1;
  const v16h* wfr0 = (const v16h*)wpack + (size_t)nt0 * KT * 32;
  const v16h* wfr1 = (const v16h*)wpack + (size_t)nt1 * KT * 32;
  const v16h* hfr  = (const v16h*)sh_hf;

  for (int step = 0; step < Lcst; ++step) {
    const int t = dir ? (Lcst - 1 - step) : step;
    // Compiler barrier: forbid hoisting the (loop-invariant) weight-fragment
    // loads out of the time loop. Without this, 24 fragments (192 VGPRs) get
    // hoisted and spilled to scratch; with it they are re-read from wpack
    // (L0/L2-hot) every step and register pressure stays ~1 iteration.
    asm volatile("" ::: "memory");
    const v16h* xfr = (const v16h*)xpack + ((size_t)bt * Lcst + t) * (KX * 32);
    v8f acc0 = {}, acc1 = {};
#pragma unroll
    for (int kt = 0; kt < KX; ++kt) {            // x part of K
      v16h a = xfr[kt * 32 + lane];
      acc0 = wmma16(a, wfr0[kt * 32 + lane], acc0);
      acc1 = wmma16(a, wfr1[kt * 32 + lane], acc1);
    }
#pragma unroll
    for (int kt = 0; kt < KH; ++kt) {            // h part of K (LDS, frag layout)
      v16h a = hfr[kt * 32 + lane];
      acc0 = wmma16(a, wfr0[(KX + kt) * 32 + lane], acc0);
      acc1 = wmma16(a, wfr1[(KX + kt) * 32 + lane], acc1);
    }
    {  // D layout: VGPR r -> row r / r+8 (lane>=16), col = lane&15
      const int col = lane & 15;
      const int mh  = (lane & 16) ? 8 : 0;
#pragma unroll
      for (int r = 0; r < 8; ++r) {
        const int row = r + mh;
        sh_g[row * G4H + nt0 * 16 + col] =
            (_Float16)(acc0[r] + bias[nt0 * 16 + col]);
        sh_g[row * G4H + nt1 * 16 + col] =
            (_Float16)(acc1[r] + bias[nt1 * 16 + col]);
      }
    }
    __syncthreads();
    // elementwise LSTM cell update; writes h back in fragment layout
#pragma unroll
    for (int e = 0; e < 4; ++e) {
      const int j = e * 1024 + tid;
      const int m = j >> 8, h = j & 255;
      const float gi = (float)sh_g[m * G4H + h];
      const float gf = (float)sh_g[m * G4H + Hcst + h];
      const float gg = (float)sh_g[m * G4H + 2 * Hcst + h];
      const float go = (float)sh_g[m * G4H + 3 * Hcst + h];
      const float c  = sigmoidf(gf) * sh_c[m * Hcst + h] +
                       sigmoidf(gi) * tanhf(gg);
      const float hv = sigmoidf(go) * tanhf(c);
      sh_c[m * Hcst + h] = c;
      // inverse fragment swizzle for element (m, h)
      const int kk = h & 31;
      const int hi = (kk >> 3) & 1;
      const int s  = ((kk & 16) >> 1) | (kk & 7);
      const int fl = (m + 16 * hi) * 16 + s;
      sh_hf[(h >> 5) * 512 + fl] = (_Float16)hv;
      hpack[((((size_t)bt * Lcst + t) * 16) + dir * 8 + (h >> 5)) * 512 + fl] =
          (_Float16)hv;
    }
    __syncthreads();
  }
}

// ---------------------------------------------------- emission = h @ W^T + b
__global__ __launch_bounds__(256) void k_cls(const _Float16* __restrict__ hpack,
                                             const _Float16* __restrict__ clspack,
                                             const float* __restrict__ bcls,
                                             float* __restrict__ emis) {
  const int wave = threadIdx.x >> 5, lane = threadIdx.x & 31;
  const int mt = blockIdx.x * 8 + wave;  // mt = bt*1024 + t  (4096 tiles)
  const v16h* afr = (const v16h*)hpack + (size_t)mt * 16 * 32;
  const v16h* bfr = (const v16h*)clspack;
  v8f acc[3] = {{}, {}, {}};
#pragma unroll 4
  for (int kt = 0; kt < 16; ++kt) {
    v16h a = afr[kt * 32 + lane];
#pragma unroll
    for (int nt = 0; nt < 3; ++nt)
      acc[nt] = wmma16(a, bfr[(nt * 16 + kt) * 32 + lane], acc[nt]);
  }
  const int b0  = (mt >> 10) * 16;
  const int t   = mt & 1023;
  const int col16 = lane & 15;
  const int mh  = (lane & 16) ? 8 : 0;
#pragma unroll
  for (int nt = 0; nt < 3; ++nt) {
    const int col = nt * 16 + col16;
    if (col < Tcst) {
#pragma unroll
      for (int r = 0; r < 8; ++r) {
        const int bb = b0 + r + mh;  // global batch index (tile base + row)
        emis[((size_t)bb * Lcst + t) * Tcst + col] = acc[nt][r] + bcls[col];
      }
    }
  }
}

// ------------------------------------------------- CRF forward + gold score
__global__ __launch_bounds__(64) void k_crf(const float* __restrict__ emis,
                                            const float* __restrict__ trans,
                                            const int* __restrict__ tags,
                                            float* __restrict__ fwd,
                                            float* __restrict__ gold) {
  const int b   = blockIdx.x;
  const int tid = threadIdx.x;
  __shared__ float str[Tcst * Tcst];
  __shared__ float fv[Tcst], nv[Tcst];
  for (int i = tid; i < Tcst * Tcst; i += 64) str[i] = trans[i];
  if (tid < Tcst) fv[tid] = (tid == START_TAG) ? 0.0f : -10000.0f;
  __syncthreads();
  for (int t = 0; t < Lcst; ++t) {
    const float* ft = emis + ((size_t)b * Lcst + t) * Tcst;
    if (tid < Tcst) {
      float mx = -3.4e38f;
      for (int p = 0; p < Tcst; ++p)
        mx = fmaxf(mx, fv[p] + str[tid * Tcst + p]);
      float s = 0.0f;
      for (int p = 0; p < Tcst; ++p)
        s += __expf(fv[p] + str[tid * Tcst + p] - mx);
      nv[tid] = mx + __logf(s) + ft[tid];
    }
    __syncthreads();
    if (tid < Tcst) fv[tid] = nv[tid];
    __syncthreads();
  }
  if (tid == 0) {
    float mx = -3.4e38f;
    for (int p = 0; p < Tcst; ++p) mx = fmaxf(mx, fv[p]);
    float s = 0.0f;
    for (int p = 0; p < Tcst; ++p) s += __expf(fv[p] - mx);
    fwd[b] = mx + __logf(s);
    int prev = START_TAG;
    float g = 0.0f;
    for (int t = 0; t < Lcst; ++t) {
      const int tg = tags[b * Lcst + t];
      g += emis[((size_t)b * Lcst + t) * Tcst + tg] + str[tg * Tcst + prev];
      prev = tg;
    }
    gold[b] = g;
  }
}

__global__ __launch_bounds__(64) void k_reduce(const float* __restrict__ fwd,
                                               const float* __restrict__ gold,
                                               float* __restrict__ out) {
  if (threadIdx.x == 0) {
    float s = 0.0f;
    for (int b = 0; b < Bcst; ++b) s += fwd[b] - gold[b];
    out[0] = s / (float)Bcst;
  }
}

extern "C" void kernel_launch(void* const* d_in, const int* in_sizes, int n_in,
                              void* d_out, int out_size, void* d_ws,
                              size_t ws_size, hipStream_t stream) {
  (void)in_sizes; (void)n_in; (void)out_size; (void)ws_size;
  const int*   batch_data = (const int*)d_in[0];
  const int*   batch_tag  = (const int*)d_in[2];
  const float* emb        = (const float*)d_in[3];
  const float* w_ih_f     = (const float*)d_in[4];
  const float* w_hh_f     = (const float*)d_in[5];
  const float* b_ih_f     = (const float*)d_in[6];
  const float* b_hh_f     = (const float*)d_in[7];
  const float* w_ih_b     = (const float*)d_in[8];
  const float* w_hh_b     = (const float*)d_in[9];
  const float* b_ih_b     = (const float*)d_in[10];
  const float* b_hh_b     = (const float*)d_in[11];
  const float* cls_w      = (const float*)d_in[12];
  const float* cls_b      = (const float*)d_in[13];
  const float* transition = (const float*)d_in[14];

  char* p = (char*)d_ws;
  auto alloc = [&](size_t bytes) -> char* {
    char* r = p;
    p += (bytes + 255) & ~(size_t)255;
    return r;
  };
  _Float16* xpack   = (_Float16*)alloc((size_t)4 * Lcst * KX * 512 * 2);
  _Float16* wpackf  = (_Float16*)alloc((size_t)NTL * KT * 512 * 2);
  _Float16* wpackb  = (_Float16*)alloc((size_t)NTL * KT * 512 * 2);
  _Float16* clspack = (_Float16*)alloc((size_t)3 * 16 * 512 * 2);
  float*    biasf   = (float*)alloc((size_t)G4H * 4);
  float*    biasb   = (float*)alloc((size_t)G4H * 4);
  _Float16* hpack   = (_Float16*)alloc((size_t)4096 * 16 * 512 * 2);
  float*    emis    = (float*)alloc((size_t)Bcst * Lcst * Tcst * 4);
  float*    fwd     = (float*)alloc((size_t)Bcst * 4);
  float*    gold    = (float*)alloc((size_t)Bcst * 4);

  k_embed_pack<<<(4 * Lcst * KX * 512) / 256, 256, 0, stream>>>(batch_data, emb,
                                                               xpack);
  k_pack_w<<<(NTL * KT * 512) / 256, 256, 0, stream>>>(
      w_ih_f, w_hh_f, b_ih_f, b_hh_f, w_ih_b, w_hh_b, b_ih_b, b_hh_b, cls_w,
      wpackf, wpackb, clspack, biasf, biasb);
  k_lstm<<<8, 1024, 0, stream>>>(xpack, wpackf, biasf, wpackb, biasb, hpack);
  k_cls<<<512, 256, 0, stream>>>(hpack, clspack, cls_b, emis);
  k_crf<<<64, 64, 0, stream>>>(emis, transition, batch_tag, fwd, gold);
  k_reduce<<<1, 64, 0, stream>>>(fwd, gold, (float*)d_out);
}
